// RetentionLayer_75041668596053
// MI455X (gfx1250) — matrix operand order repaired and checked by reading, not compiled, not run on hardware
//
#include <hip/hip_runtime.h>
#include <hip/hip_bf16.h>

// ---------------------------------------------------------------------------
// RetentionLayer forward for MI455X (gfx1250, wave32, WMMA).
// B=2, T=2048, D=1024, H=16, HD=64.
// ---------------------------------------------------------------------------

#define B_  2
#define T_  2048
#define D_  1024
#define H_  16
#define HD_ 64
#define MT_ (B_ * T_)          // 4096 rows
#define EPS_ 1e-6f

typedef __bf16 bf16_t;
typedef __attribute__((ext_vector_type(16))) __bf16 v16bf;
typedef __attribute__((ext_vector_type(8)))  __bf16 v8bf;
typedef __attribute__((ext_vector_type(4)))  __bf16 v4bf;
typedef __attribute__((ext_vector_type(8)))  float  v8f;
typedef __attribute__((ext_vector_type(4)))  float  v4f;

__device__ __forceinline__ v8f v8f_zero() {
    v8f z;
#pragma unroll
    for (int i = 0; i < 8; ++i) z[i] = 0.f;
    return z;
}

// WMMA D = A(16x32 bf16) * B(32x16 bf16) + C(16x16 f32)
__device__ __forceinline__ v8f wmma_bf16(v16bf a, v16bf b, v8f c) {
    return __builtin_amdgcn_wmma_f32_16x16x32_bf16(false, a, false, b,
                                                   (short)0, c, false, false);
}

// A-operand (16x32, row-major source, leading dim ldk):
// lane m = lane&15 is the matrix row; half = lane>>4 selects K chunks
// {8h..8h+7} and {16+8h..16+8h+7} (ISA 16-bit A table).
__device__ __forceinline__ v16bf load_a16(const bf16_t* base, int ldk, int lane) {
    const bf16_t* p = base + (size_t)(lane & 15) * ldk + ((lane >> 4) << 3);
    v8bf c0 = *(const v8bf*)(p);
    v8bf c1 = *(const v8bf*)(p + 16);
    v16bf r;
#pragma unroll
    for (int i = 0; i < 8; ++i) { r[i] = c0[i]; r[i + 8] = c1[i]; }
    return r;
}

// B-operand (32x16): lane n = lane&15 is the matrix column; half = lane>>4
// selects K chunks {8h..8h+7} and {16+8h..16+8h+7} (matches A table).
// Source rows are B^T rows (i.e. weight rows / V^T rows), leading dim ldk.
__device__ __forceinline__ v16bf load_b16(const bf16_t* base, int ldk, int lane) {
    const bf16_t* p = base + (size_t)(lane & 15) * ldk + ((lane >> 4) << 3);
    v8bf c0 = *(const v8bf*)(p);
    v8bf c1 = *(const v8bf*)(p + 16);
    v16bf r;
#pragma unroll
    for (int i = 0; i < 8; ++i) { r[i] = c0[i]; r[i + 8] = c1[i]; }
    return r;
}

// Async Global -> LDS direct copy (ASYNCcnt path, no VGPR transit).
// lds_addr is the wave-relative LDS byte address (low 32 bits of the flat
// shared-memory pointer, per the LDS aperture truncation rule).
__device__ __forceinline__ void async_g2lds_b128(unsigned int lds_addr,
                                                 unsigned long long gaddr) {
    asm volatile("global_load_async_to_lds_b128 %0, %1, off"
                 :: "v"(lds_addr), "v"(gaddr) : "memory");
}
__device__ __forceinline__ void wait_asynccnt0() {
    asm volatile("s_wait_asynccnt 0" ::: "memory");
}

// ---------------------------------------------------------------------------
// 1) RMSNorm -> bf16   (one block per token row)
// ---------------------------------------------------------------------------
__global__ void rmsnorm_bf16_kernel(const float* __restrict__ x,
                                    const float* __restrict__ nw,
                                    bf16_t* __restrict__ xn) {
    const int row = blockIdx.x;
    const int tid = threadIdx.x;                 // 128 threads
    const float* xr = x + (size_t)row * D_;
    float vals[8];
    float ss = 0.f;
#pragma unroll
    for (int i = 0; i < 8; ++i) {
        float v = xr[tid + i * 128];
        vals[i] = v;
        ss += v * v;
    }
    __shared__ float red[128];
    red[tid] = ss;
    __syncthreads();
#pragma unroll
    for (int s = 64; s > 0; s >>= 1) {
        if (tid < s) red[tid] += red[tid + s];
        __syncthreads();
    }
    const float rn = rsqrtf(red[0] * (1.0f / D_) + EPS_);
#pragma unroll
    for (int i = 0; i < 8; ++i) {
        int c = tid + i * 128;
        xn[(size_t)row * D_ + c] = (bf16_t)(vals[i] * rn * nw[c]);
    }
}

// ---------------------------------------------------------------------------
// 2) f32 -> bf16 weight conversion (n multiple of 1024)
// ---------------------------------------------------------------------------
__global__ void f32_to_bf16_kernel(const float* __restrict__ in,
                                   bf16_t* __restrict__ out, int n) {
    int i = (blockIdx.x * blockDim.x + threadIdx.x) * 4;
    if (i < n) {
        v4f v = *(const v4f*)(in + i);
        v4bf o;
#pragma unroll
        for (int r = 0; r < 4; ++r) o[r] = (bf16_t)v[r];
        *(v4bf*)(out + i) = o;
    }
}

// ---------------------------------------------------------------------------
// 3) Tiled WMMA GEMM: C[M,N] = A[M,K] @ W[N,K]^T, M=4096, N=K=1024.
//    Block: 128 threads = 4 waves; block tile 64(M) x 64(N);
//    wave tile 16(M) x 64(N). W-rows staged into LDS with async
//    global->LDS copies (ASYNCcnt), shared by the 4 waves.
//    mode 0: bf16 row-major out          (Q, K projections)
//    mode 1: bf16 per-head transposed out [B,H,HD,T]  (V projection)
//    mode 2: f32  out = resid + acc*ls[n] (output projection epilogue)
// ---------------------------------------------------------------------------
__global__ void gemm_wmma_kernel(const bf16_t* __restrict__ A,
                                 const bf16_t* __restrict__ W,
                                 bf16_t* __restrict__ Cb,
                                 float* __restrict__ Cf,
                                 const float* __restrict__ resid,
                                 const float* __restrict__ ls,
                                 int mode) {
    __shared__ bf16_t sB[64 * 32];               // 64 W-rows x 32 K, 4 KB
    const int n0   = blockIdx.x * 64;
    const int m0   = blockIdx.y * 64;
    const int tid  = threadIdx.x;
    const int wave = tid >> 5;
    const int lane = tid & 31;
    const int mw   = m0 + wave * 16;

    v8f acc[4];
#pragma unroll
    for (int i = 0; i < 4; ++i) acc[i] = v8f_zero();

    const int sr = tid >> 1;                     // 0..63 : W row within strip
    const int sh = (tid & 1) << 4;               // 0 / 16 : K half
    const bf16_t* wsrc = W + (size_t)(n0 + sr) * D_;
    const bf16_t* arow = A + (size_t)mw * D_;
    const unsigned int lds_dst =
        (unsigned int)(size_t)(&sB[sr * 32 + sh]);   // wave-relative LDS addr

    for (int k0 = 0; k0 < D_; k0 += 32) {
        // async stage of the shared B strip straight into LDS (32 B / thread)
        async_g2lds_b128(lds_dst,
                         (unsigned long long)(size_t)(wsrc + k0 + sh));
        async_g2lds_b128(lds_dst + 16,
                         (unsigned long long)(size_t)(wsrc + k0 + sh + 8));
        __builtin_prefetch(arow + k0 + 64, 0, 1);   // global_prefetch next A
        wait_asynccnt0();
        __syncthreads();

        v16bf a = load_a16(arow + k0, D_, lane);
#pragma unroll
        for (int nt = 0; nt < 4; ++nt) {
            v16bf b = load_b16(sB + nt * 16 * 32, 32, lane);
            acc[nt] = wmma_bf16(a, b, acc[nt]);
        }
        __syncthreads();
    }

    // C/D VGPR layout: lane l, vgpr r -> M = r + 8*(l>>4), N = l&15
    const int col   = lane & 15;
    const int rbase = (lane >> 4) << 3;

    if (mode == 0) {
#pragma unroll
        for (int nt = 0; nt < 4; ++nt)
#pragma unroll
            for (int r = 0; r < 8; ++r)
                Cb[(size_t)(mw + rbase + r) * D_ + n0 + nt * 16 + col] =
                    (bf16_t)acc[nt][r];
    } else if (mode == 1) {
        const int bb = mw / T_;
        const int t0 = (mw - bb * T_) + rbase;   // 8 consecutive t values
#pragma unroll
        for (int nt = 0; nt < 4; ++nt) {
            int n  = n0 + nt * 16 + col;
            int h  = n >> 6;
            int hd = n & 63;
            v8bf pk;
#pragma unroll
            for (int r = 0; r < 8; ++r) pk[r] = (bf16_t)acc[nt][r];
            *(v8bf*)&Cb[(((size_t)bb * H_ + h) * HD_ + hd) * T_ + t0] = pk;
        }
    } else {
#pragma unroll
        for (int nt = 0; nt < 4; ++nt)
#pragma unroll
            for (int r = 0; r < 8; ++r) {
                int n = n0 + nt * 16 + col;
                size_t idx = (size_t)(mw + rbase + r) * D_ + n;
                Cf[idx] = resid[idx] + acc[nt][r] * ls[n];
            }
    }
}

// ---------------------------------------------------------------------------
// 4) Retention attention: one wave per (b, h, 16-row query tile).
//    For each 32-wide key chunk:  S = Q K^T (2x2 WMMAs), apply causal decay
//    scale*g^(i-j) on the f32 accumulator, re-lay out S as an A-operand via
//    LDS, then out += S @ V (4 WMMAs against V^T tiles).
// ---------------------------------------------------------------------------
__global__ void retention_attn_kernel(const bf16_t* __restrict__ Q,
                                      const bf16_t* __restrict__ K,
                                      const bf16_t* __restrict__ Vt,
                                      const float* __restrict__ gamma,
                                      bf16_t* __restrict__ AO) {
    __shared__ bf16_t sS[16 * 32];               // decayed score tile, 1 KB
    const int q0   = blockIdx.x * 16;
    const int h    = blockIdx.y;
    const int b    = blockIdx.z;
    const int lane = threadIdx.x;

    const bf16_t* qb = Q + ((size_t)b * T_ + q0) * D_ + h * HD_;
    const bf16_t* kb = K + (size_t)b * T_ * D_ + h * HD_;
    const bf16_t* vb = Vt + ((size_t)b * H_ + h) * (size_t)HD_ * T_;

    const float g     = fminf(fmaxf(gamma[h], 0.01f), 0.999f);
    const float l2g   = log2f(g);
    const float scale = 0.125f;                  // HD^-0.5

    // Q operand is loop-invariant: 16 rows x 64 d, two A tiles
    v16bf qa0 = load_a16(qb, D_, lane);
    v16bf qa1 = load_a16(qb + 32, D_, lane);

    v8f acc[4];
#pragma unroll
    for (int i = 0; i < 4; ++i) acc[i] = v8f_zero();

    const int col   = lane & 15;
    const int rbase = (lane >> 4) << 3;
    const int nchunks = q0 / 32 + 1;             // covers k in [0, q0+16)

    for (int jc = 0; jc < nchunks; ++jc) {
        const int k0 = jc * 32;
#pragma unroll
        for (int sub = 0; sub < 2; ++sub) {
            const int kk = k0 + sub * 16;
            v16bf kt0 = load_b16(kb + (size_t)kk * D_, D_, lane);
            v16bf kt1 = load_b16(kb + (size_t)kk * D_ + 32, D_, lane);
            v8f s = v8f_zero();
            s = wmma_bf16(qa0, kt0, s);
            s = wmma_bf16(qa1, kt1, s);
            // causal decay applied directly on the C-layout accumulator
#pragma unroll
            for (int r = 0; r < 8; ++r) {
                int qi = q0 + rbase + r;
                int ki = kk + col;
                float val = (qi >= ki)
                          ? s[r] * scale * exp2f((float)(qi - ki) * l2g)
                          : 0.f;
                sS[(rbase + r) * 32 + sub * 16 + col] = (bf16_t)val;
            }
        }
        __syncthreads();                         // fence LDS write -> read
        v16bf sa = load_a16(sS, 32, lane);       // score tile as A-operand
        __syncthreads();
#pragma unroll
        for (int nt = 0; nt < 4; ++nt) {
            v16bf vt = load_b16(vb + (size_t)(nt * 16) * T_ + k0, T_, lane);
            acc[nt] = wmma_bf16(sa, vt, acc[nt]);
        }
    }

    bf16_t* ao = AO + ((size_t)b * T_ + q0) * D_ + h * HD_;
#pragma unroll
    for (int nt = 0; nt < 4; ++nt)
#pragma unroll
        for (int r = 0; r < 8; ++r)
            ao[(size_t)(rbase + r) * D_ + nt * 16 + col] = (bf16_t)acc[nt][r];
}

// ---------------------------------------------------------------------------
// Launch: rmsnorm -> weight cvt -> Q/K/V GEMMs -> retention -> O GEMM+epilogue
// ---------------------------------------------------------------------------
extern "C" void kernel_launch(void* const* d_in, const int* in_sizes, int n_in,
                              void* d_out, int out_size, void* d_ws, size_t ws_size,
                              hipStream_t stream) {
    const float* x        = (const float*)d_in[0];
    const float* wq       = (const float*)d_in[1];
    const float* wk       = (const float*)d_in[2];
    const float* wv       = (const float*)d_in[3];
    const float* wo       = (const float*)d_in[4];
    const float* gamma    = (const float*)d_in[5];
    const float* norm_w   = (const float*)d_in[6];
    const float* ls_gamma = (const float*)d_in[7];
    float* out = (float*)d_out;

    bf16_t* ws  = (bf16_t*)d_ws;
    const size_t SZ_ACT = (size_t)MT_ * D_;      // 4M elems
    const size_t SZ_W   = (size_t)D_ * D_;       // 1M elems
    bf16_t* xn  = ws;                            // normalized activations
    bf16_t* wqb = xn  + SZ_ACT;
    bf16_t* wkb = wqb + SZ_W;
    bf16_t* wvb = wkb + SZ_W;
    bf16_t* wob = wvb + SZ_W;
    bf16_t* Qb  = wob + SZ_W;                    // [B*T, D] bf16
    bf16_t* Kb  = Qb  + SZ_ACT;                  // [B*T, D] bf16
    bf16_t* Vt  = Kb  + SZ_ACT;                  // [B, H, HD, T] bf16
    bf16_t* AO  = Vt  + SZ_ACT;                  // attention out [B*T, D] bf16

    rmsnorm_bf16_kernel<<<dim3(MT_), 128, 0, stream>>>(x, norm_w, xn);

    f32_to_bf16_kernel<<<dim3(1024), 256, 0, stream>>>(wq, wqb, (int)SZ_W);
    f32_to_bf16_kernel<<<dim3(1024), 256, 0, stream>>>(wk, wkb, (int)SZ_W);
    f32_to_bf16_kernel<<<dim3(1024), 256, 0, stream>>>(wv, wvb, (int)SZ_W);
    f32_to_bf16_kernel<<<dim3(1024), 256, 0, stream>>>(wo, wob, (int)SZ_W);

    dim3 ggrid(D_ / 64, MT_ / 64);               // (16, 64)
    gemm_wmma_kernel<<<ggrid, 128, 0, stream>>>(xn, wqb, Qb, nullptr, nullptr, nullptr, 0);
    gemm_wmma_kernel<<<ggrid, 128, 0, stream>>>(xn, wkb, Kb, nullptr, nullptr, nullptr, 0);
    gemm_wmma_kernel<<<ggrid, 128, 0, stream>>>(xn, wvb, Vt, nullptr, nullptr, nullptr, 1);

    retention_attn_kernel<<<dim3(T_ / 16, H_, B_), 32, 0, stream>>>(Qb, Kb, Vt, gamma, AO);

    gemm_wmma_kernel<<<ggrid, 128, 0, stream>>>(AO, wob, nullptr, out, x, ls_gamma, 2);
}
